// Reservoir_77094662963394
// MI455X (gfx1250) — compile-verified, matrix operand order checked by reading, not run on hardware
//
#include <hip/hip_runtime.h>
#include <hip/hip_bf16.h>
#include <math.h>

// Problem constants (from reference)
#define BATCH 32
#define TT    2048
#define INF   128
#define RESV  1024
#define PAD   8          // bf16 elements of row padding in LDS (16B) to stagger banks

typedef __attribute__((ext_vector_type(16))) __bf16        v16bf;
typedef __attribute__((ext_vector_type(8)))  float         v8f;
typedef __attribute__((ext_vector_type(4)))  unsigned int  uint4v;
typedef __attribute__((ext_vector_type(4)))  int           v4i;

#if defined(__has_builtin)
#if __has_builtin(__builtin_amdgcn_global_load_async_to_lds_b128)
#define USE_ASYNC_LDS 1
#endif
#endif

union Frag { uint4v u[2]; v16bf v; };

// Load a 16-element bf16 fragment (32 bytes) from LDS as two ds_load_b128.
__device__ __forceinline__ v16bf ld_frag(const __bf16* p) {
    Frag f;
    f.u[0] = *(const uint4v*)(p);
    f.u[1] = *(const uint4v*)(p + 8);
    return f.v;
}

// Load 16 consecutive f32 from global and round to bf16 (v_cvt_pk_bf16_f32).
__device__ __forceinline__ v16bf ld_cvt16(const float* __restrict__ p) {
    v16bf r;
#pragma unroll
    for (int i = 0; i < 16; ++i) r[i] = (__bf16)p[i];
    return r;
}

__global__ void init_counter(unsigned* c) { *c = 0u; }

// ---------------------------------------------------------------------------
// Phase 1: proj[m, n] = sum_k x[m, k] * W_in[n, k], m = b*T + t flattened.
// Grid: (RESV/16, (BATCH*TT)/32), block = 64 threads (2 waves).
// Each wave owns a 16x16 output tile; K = 128 -> 4 x v_wmma_f32_16x16x32_bf16.
// Memory-bound phase (~290 MB traffic -> ~13 us at 23.3 TB/s); bf16 WMMA keeps
// compute far under the memory roof (fp32 VALU would not).
// ---------------------------------------------------------------------------
__global__ void proj_wmma(const float* __restrict__ x,
                          const float* __restrict__ Win,
                          float* __restrict__ out) {
    const int lane = threadIdx.x & 31;
    const int wave = threadIdx.x >> 5;
    const int n0   = blockIdx.x * 16;
    const int m0   = blockIdx.y * 32;
    const int nl   = lane & 15;
    const int kh   = (lane >> 4) * 16;          // K half selected by lane group
    const int arow = m0 + wave * 16 + nl;       // A-matrix row for this lane

    v8f c = {};
#pragma unroll
    for (int k0 = 0; k0 < INF; k0 += 32) {
        v16bf a = ld_cvt16(x   + (size_t)arow      * INF + k0 + kh);
        v16bf b = ld_cvt16(Win + (size_t)(n0 + nl) * INF + k0 + kh);
        c = __builtin_amdgcn_wmma_f32_16x16x32_bf16(false, a, false, b,
                                                    (short)0, c, false, false);
    }
    // C/D layout: VGPR j -> M = j (lanes 0-15) / j+8 (lanes 16-31), N = lane%16
    const int mb = m0 + wave * 16 + ((lane >> 4) << 3);
#pragma unroll
    for (int j = 0; j < 8; ++j)
        out[(size_t)(mb + j) * RESV + n0 + nl] = c[j];
}

// ---------------------------------------------------------------------------
// Phase 2: persistent scan (latency-bound: 2048 dependent steps).
// 64 WGs; each pins a 16-column slice of W_res in LDS as bf16 (32 KB) and owns
// the matching 32x16 output tile. Per step: async-copy S_{t-1} (64 KB bf16,
// L2-resident ping-pong) into LDS, 32 x v_wmma_f32_16x16x32_bf16 per wave
// (K=1024), tanh, write f32 state over proj in d_out + bf16 state to the other
// ping-pong buffer, then one global-atomic arrive per WG + acquire spin.
// ---------------------------------------------------------------------------
__global__ void reservoir_scan(const float* __restrict__ Wres,
                               float* __restrict__ out,      // proj in, states out
                               __bf16* __restrict__ Sbuf,    // 2 x [32 x 1024] bf16
                               unsigned* __restrict__ counter) {
    __shared__ __attribute__((aligned(16))) __bf16 Wlds[16    * (RESV + PAD)];
    __shared__ __attribute__((aligned(16))) __bf16 Slds[BATCH * (RESV + PAD)];

    const int tid  = threadIdx.x;       // 0..63
    const int lane = tid & 31;
    const int wave = tid >> 5;
    const int n0   = blockIdx.x * 16;
    const int nl   = lane & 15;
    const int kh   = (lane >> 4) * 16;
    const int arow = wave * 16 + nl;    // state row (batch index) for A frags
    const unsigned nwg = gridDim.x;

    // One-time: pin this WG's W_res slice (rows n0..n0+15, bf16) in LDS.
    for (int idx = tid; idx < 16 * RESV; idx += 64) {
        const int r = idx >> 10, k = idx & (RESV - 1);
        Wlds[r * (RESV + PAD) + k] = (__bf16)Wres[(size_t)(n0 + r) * RESV + k];
    }
    __syncthreads();

    for (int t = 0; t < TT; ++t) {
        v8f c = {};
        if (t > 0) {
            // Stage S_{t-1} from the L2-resident ping-pong buffer into LDS.
            const __bf16* src = Sbuf + (size_t)((t + 1) & 1) * (BATCH * RESV);
#if defined(USE_ASYNC_LDS)
            // CDNA5 async path: global -> LDS without VGPR round trip
            // (GLOBAL_LOAD_ASYNC_TO_LDS_B128, tracked by ASYNCcnt).
            for (int idx = tid; idx < (BATCH * RESV) / 8; idx += 64) {
                const int e = idx * 8;
                const int r = e >> 10, k = e & (RESV - 1);
                __builtin_amdgcn_global_load_async_to_lds_b128(
                    (v4i*)(src + e),
                    (v4i*)(&Slds[r * (RESV + PAD) + k]),
                    0, 0);
            }
#if defined(__has_builtin) && __has_builtin(__builtin_amdgcn_s_wait_asynccnt)
            __builtin_amdgcn_s_wait_asynccnt(0);
#else
            asm volatile("s_wait_asynccnt 0x0" ::: "memory");
#endif
#else
            for (int idx = tid; idx < (BATCH * RESV) / 8; idx += 64) {
                const int e = idx * 8;
                const int r = e >> 10, k = e & (RESV - 1);
                *(uint4v*)(&Slds[r * (RESV + PAD) + k]) = *(const uint4v*)(src + e);
            }
#endif
            __syncthreads();

#pragma unroll 4
            for (int k0 = 0; k0 < RESV; k0 += 32) {
                v16bf a = ld_frag(&Slds[arow * (RESV + PAD) + k0 + kh]);
                v16bf b = ld_frag(&Wlds[nl   * (RESV + PAD) + k0 + kh]);
                c = __builtin_amdgcn_wmma_f32_16x16x32_bf16(false, a, false, b,
                                                            (short)0, c, false, false);
            }
        }

        // s = tanh(proj_t + S @ W^T); write f32 state over proj in d_out and
        // bf16 state into the other ping-pong buffer.
        const int mb = wave * 16 + ((lane >> 4) << 3);
        __bf16* dst = Sbuf + (size_t)(t & 1) * (BATCH * RESV);
#pragma unroll
        for (int j = 0; j < 8; ++j) {
            const int b_idx = mb + j;                         // batch index
            const size_t oa = ((size_t)b_idx * TT + t) * RESV + n0 + nl;
            const float s = tanhf(out[oa] + c[j]);
            out[oa] = s;
            dst[b_idx * RESV + n0 + nl] = (__bf16)s;
        }

        // Device-wide step barrier: release state, arrive, acquire-spin.
        __threadfence();
        __syncthreads();
        if (tid == 0) {
            atomicAdd(counter, 1u);
            const unsigned target = nwg * (unsigned)(t + 1);
            while (__hip_atomic_load(counter, __ATOMIC_ACQUIRE,
                                     __HIP_MEMORY_SCOPE_AGENT) < target)
                __builtin_amdgcn_s_sleep(2);
        }
        __syncthreads();
    }
}

extern "C" void kernel_launch(void* const* d_in, const int* in_sizes, int n_in,
                              void* d_out, int out_size, void* d_ws, size_t ws_size,
                              hipStream_t stream) {
    const float* x    = (const float*)d_in[0];   // [32, 2048, 128]
    const float* Win  = (const float*)d_in[1];   // [1024, 128]
    const float* Wres = (const float*)d_in[2];   // [1024, 1024]
    float* out        = (float*)d_out;           // [32, 2048, 1024]

    unsigned* counter = (unsigned*)d_ws;
    __bf16*   Sbuf    = (__bf16*)((char*)d_ws + 256);  // 2 x 64 KB ping-pong

    init_counter<<<1, 1, 0, stream>>>(counter);

    proj_wmma<<<dim3(RESV / 16, (BATCH * TT) / 32), 64, 0, stream>>>(x, Win, out);

    reservoir_scan<<<dim3(RESV / 16), 64, 0, stream>>>(Wres, out, Sbuf, counter);
}
